// conv3d_sa_dgcnn_encoder_36352603193509
// MI455X (gfx1250) — matrix operand order repaired and checked by reading, not compiled, not run on hardware
//
#include <hip/hip_runtime.h>

// ---------------------------------------------------------------------------
// MI455X (gfx1250, wave32) implementation.
// Heavy ops (conv1 implicit GEMM, conv2/conv5 1x1 convs, stage-2 attention
// matmuls) run on v_wmma_f32_16x16x32_bf16 (bf16 in, f32 accumulate).
// Tiny stage-1 attention (C=3), softmax/top-k and max reductions are scalar f32.
// ---------------------------------------------------------------------------

typedef __attribute__((ext_vector_type(16))) unsigned short v16u;
typedef __attribute__((ext_vector_type(8)))  unsigned short v8u;
typedef __attribute__((ext_vector_type(8)))  float          v8f;
typedef __attribute__((ext_vector_type(16))) __bf16         v16bf;

__device__ __forceinline__ unsigned short f2bf(float x) {
  unsigned u = __builtin_bit_cast(unsigned, x);
  unsigned r = u + 0x7FFFu + ((u >> 16) & 1u);   // round-to-nearest-even
  return (unsigned short)(r >> 16);
}
__device__ __forceinline__ float bf2f(unsigned short h) {
  unsigned u = ((unsigned)h) << 16;
  return __builtin_bit_cast(float, u);
}

// load two 8-half (16B) runs into one 16-half fragment register
__device__ __forceinline__ v16u ld_frag2(const unsigned short* p0,
                                         const unsigned short* p1) {
  v16u r;
  *reinterpret_cast<v8u*>(&r)       = *reinterpret_cast<const v8u*>(p0);
  *(reinterpret_cast<v8u*>(&r) + 1) = *reinterpret_cast<const v8u*>(p1);
  return r;
}

__device__ __forceinline__ v8f wmma_bf16(v16u a, v16u b, v8f c) {
  union U { v16u u; v16bf h; };
  U A, B; A.u = a; B.u = b;
  return __builtin_amdgcn_wmma_f32_16x16x32_bf16(
      false, A.h, false, B.h, (short)0, c, false, false);
}

#define P1TOT 293760   // 8*8*255*18 conv1 output pixels
#define P2TOT 326400   // 64*255*20  conv2 output pixels
#define P5TOT 16320    // 64*255     conv5 output pixels

// ---------------------------------------------------------------------------
// Weight repack: fp32 [M][K] row-major -> bf16 A-fragment order
// dst[((mt*KC+kc)*32+lane)*16 + j], lane<16: row=mt*16+lane, run {0..15}
// ---------------------------------------------------------------------------
__global__ void k_pack(const float* __restrict__ src, unsigned short* __restrict__ dst,
                       int M, int K, int KC) {
  int gid = blockIdx.x * blockDim.x + threadIdx.x;
  int total = (M / 16) * KC * 32;
  if (gid >= total) return;
  int lane = gid & 31;
  int rest = gid >> 5;
  int kc = rest % KC, mt = rest / KC;
  int hi = lane >> 4;
  int m = mt * 16 + (lane & 15);
  unsigned short* p = dst + (size_t)gid * 16;
#pragma unroll
  for (int j = 0; j < 16; ++j) {
    int kl = j + (hi ? 16 : 0);
    int k = kc * 32 + kl;
    p[j] = (k < K) ? f2bf(src[(size_t)m * K + k]) : (unsigned short)0;
  }
}

// ---------------------------------------------------------------------------
// Stage-1: per-group (G=256, N=512, C=3) attention + edge features -> y1 bf16
// y1 layout (B,6,32,512,20): idx = (((b*6+c)*32+f)*512+n)*20+kk
// ---------------------------------------------------------------------------
__global__ __launch_bounds__(256) void k_stage1(
    const float* __restrict__ x, const float* __restrict__ Wq,
    const float* __restrict__ Wk, const float* __restrict__ Wv,
    unsigned short* __restrict__ y1) {
  __shared__ float sk[512][3], sv[512][3], sx[512][3], sxt[512][3];
  __shared__ short sidx1[512][20];
  int g = blockIdx.x, b = g >> 5, f = g & 31;
  const float* xg = x + (size_t)(b * 32 + f) * 3 * 512;  // xg[c*512+n]
  int tid = threadIdx.x;
  float wq[9], wk[9], wv[9];
#pragma unroll
  for (int i = 0; i < 9; ++i) { wq[i] = Wq[i]; wk[i] = Wk[i]; wv[i] = Wv[i]; }
  for (int n = tid; n < 512; n += 256) {
    float c0 = xg[n], c1 = xg[512 + n], c2 = xg[1024 + n];
    sx[n][0] = c0; sx[n][1] = c1; sx[n][2] = c2;
#pragma unroll
    for (int i = 0; i < 3; ++i) {
      sk[n][i] = wk[i*3]*c0 + wk[i*3+1]*c1 + wk[i*3+2]*c2;
      sv[n][i] = wv[i*3]*c0 + wv[i*3+1]*c1 + wv[i*3+2]*c2;
    }
  }
  __syncthreads();
  const float scale = 0.57735026919f;  // 1/sqrt(3)
  for (int n = tid; n < 512; n += 256) {
    float q0 = wq[0]*sx[n][0] + wq[1]*sx[n][1] + wq[2]*sx[n][2];
    float q1 = wq[3]*sx[n][0] + wq[4]*sx[n][1] + wq[5]*sx[n][2];
    float q2 = wq[6]*sx[n][0] + wq[7]*sx[n][1] + wq[8]*sx[n][2];
    float mrun = -3.4e38f, ssum = 0.f, o0 = 0.f, o1 = 0.f, o2 = 0.f;
    float tv[20]; short ti[20];
#pragma unroll
    for (int j = 0; j < 20; ++j) { tv[j] = -3.4e38f; ti[j] = 0; }
    for (int m = 0; m < 512; ++m) {
      float l = (q0*sk[m][0] + q1*sk[m][1] + q2*sk[m][2]) * scale;
      if (l > mrun) {
        float sc = __expf(mrun - l);
        ssum *= sc; o0 *= sc; o1 *= sc; o2 *= sc; mrun = l;
      }
      float p = __expf(l - mrun);
      ssum += p; o0 += p*sv[m][0]; o1 += p*sv[m][1]; o2 += p*sv[m][2];
      if (l > tv[19]) {  // descending order, ties keep lower index (stable)
        int pos = 19;
        while (pos > 0 && l > tv[pos-1]) { tv[pos] = tv[pos-1]; ti[pos] = ti[pos-1]; --pos; }
        tv[pos] = l; ti[pos] = (short)m;
      }
    }
    float inv = 1.f / ssum;
    sxt[n][0] = sx[n][0] + o0*inv;
    sxt[n][1] = sx[n][1] + o1*inv;
    sxt[n][2] = sx[n][2] + o2*inv;
#pragma unroll
    for (int j = 0; j < 20; ++j) sidx1[n][j] = ti[j];
  }
  __syncthreads();
  const size_t CH = 32u * 512u * 20u;  // channel stride in y1
  for (int n = tid; n < 512; n += 256) {
    float c0 = sxt[n][0], c1 = sxt[n][1], c2 = sxt[n][2];
    size_t base = ((((size_t)b * 6) * 32 + f) * 512 + n) * 20;
    for (int kk = 0; kk < 20; ++kk) {
      int j = sidx1[n][kk];
      y1[base + kk + 0*CH] = f2bf(sxt[j][0] - c0);
      y1[base + kk + 1*CH] = f2bf(sxt[j][1] - c1);
      y1[base + kk + 2*CH] = f2bf(sxt[j][2] - c2);
      y1[base + kk + 3*CH] = f2bf(c0);
      y1[base + kk + 4*CH] = f2bf(c1);
      y1[base + kk + 5*CH] = f2bf(c2);
    }
  }
}

// ---------------------------------------------------------------------------
// conv1: implicit GEMM, M=64 ch, K=216 (pad 224), 64-pixel tiles. WMMA bf16.
// Epilogue: bn1 + LeakyReLU. act1[o][P], P=((b*8+f')*255+n')*18+k'
// ---------------------------------------------------------------------------
__global__ __launch_bounds__(128) void k_conv1(
    const unsigned short* __restrict__ y1, const unsigned short* __restrict__ Wp,
    const float* __restrict__ g1, const float* __restrict__ b1,
    float* __restrict__ act1) {
  __shared__ __align__(16) unsigned short patch[64][224];  // [pixel][t] bf16
  int tid = threadIdx.x;
  int Pb = blockIdx.x * 64;
  {
    int pl = tid >> 1, half = tid & 1;
    int P = Pb + pl;
    int kp = P % 18; int r = P / 18;
    int np = r % 255; r /= 255;
    int fp = r & 7, b = r >> 3;
    for (int t = half * 112; t < half * 112 + 112; ++t) {
      unsigned short v = 0;
      if (t < 216) {
        int c = t / 36; int rem = t - c * 36;
        int dd = rem / 9; rem -= dd * 9;
        int hh = rem / 3; int ww = rem - hh * 3;
        size_t idx = ((((size_t)b*6 + c)*32 + (4*fp+dd))*512 + (2*np+hh))*20 + (kp+ww);
        v = y1[idx];
      }
      patch[pl][t] = v;
    }
  }
  __syncthreads();
  int wave = tid >> 5, lane = tid & 31, lr = lane & 15, hi = lane >> 4;
  int mt = wave;
  float rs = rsqrtf(1.00001f);
  for (int nt = 0; nt < 4; ++nt) {
    v8f acc = {};
#pragma unroll
    for (int kc = 0; kc < 7; ++kc) {
      const unsigned short* pa = Wp + (size_t)((mt*7 + kc)*32 + lane) * 16;
      v16u a = ld_frag2(pa, pa + 8);                       // packed A
      const unsigned short* pb = &patch[nt*16 + lr][kc*32 + (hi ? 16 : 0)];
      v16u bb = ld_frag2(pb, pb + 8);                      // [N][K] B
      acc = wmma_bf16(a, bb, acc);
    }
    int p = Pb + nt*16 + lr;
#pragma unroll
    for (int r = 0; r < 8; ++r) {
      int o = mt*16 + r + (hi ? 8 : 0);
      float val = acc[r] * (g1[o] * rs) + b1[o];
      val = val >= 0.f ? val : 0.2f * val;
      act1[(size_t)o * P1TOT + p] = val;
    }
  }
}

// ---------------------------------------------------------------------------
// max over W'=18 -> x1; emit xbf[g2][256 pad][64] bf16 and cat[p2][192] (c<64)
// ---------------------------------------------------------------------------
__global__ void k_max1(const float* __restrict__ act1,
                       unsigned short* __restrict__ xbf,
                       unsigned short* __restrict__ catb) {
  int gid = blockIdx.x * 256 + threadIdx.x;
  if (gid < 64 * 64) {  // zero pad row n=255 of each group
    int g2 = gid >> 6, c = gid & 63;
    xbf[((size_t)g2 * 256 + 255) * 64 + c] = 0;
  }
  if (gid >= 64 * 8 * 8 * 255) return;
  int n = gid % 255; int t = gid / 255;
  int f = t & 7; t >>= 3; int b = t & 7; int c = t >> 3;
  int g2 = b * 8 + f;
  size_t base = (size_t)c * P1TOT + ((size_t)(g2 * 255) + n) * 18;
  float m = -3.4e38f;
#pragma unroll
  for (int j = 0; j < 18; ++j) m = fmaxf(m, act1[base + j]);
  unsigned short h = f2bf(m);
  xbf[((size_t)g2 * 256 + n) * 64 + c] = h;
  catb[((size_t)g2 * 255 + n) * 192 + c] = h;
}

// ---------------------------------------------------------------------------
// Stage-2: per-group (G2=64, N=255 pad 256, C=64) attention, all matmuls WMMA.
// Dynamic LDS: sx/qA/kB/vT/xt (bf16) + probs + logits + top-k indices.
// Output y2[g2][n][kk][c128] bf16 (edge features, B-operand layout for conv2).
// ---------------------------------------------------------------------------
__global__ __launch_bounds__(256) void k_stage2(
    const unsigned short* __restrict__ xbf,
    const unsigned short* __restrict__ Wqp, const unsigned short* __restrict__ Wkp,
    const unsigned short* __restrict__ Wvp, unsigned short* __restrict__ y2) {
  extern __shared__ __align__(32) char smem[];
  unsigned short* sx = (unsigned short*)(smem);            // [256][64]  x bf16
  unsigned short* qA = (unsigned short*)(smem + 32768);    // [256][64]  q rows (A)
  unsigned short* kB = (unsigned short*)(smem + 65536);    // [256][64]  k (B: [m][c])
  unsigned short* vT = (unsigned short*)(smem + 98304);    // [64][256]  v (B: [c][m])
  unsigned short* xt = (unsigned short*)(smem + 131072);   // [256][64]  x + attn out
  unsigned short* pr = (unsigned short*)(smem + 163840);   // [16][256]  probs (A)
  float*          lg = (float*)(smem + 172032);            // [16][256]  logits
  short*        sidx = (short*)(smem + 188416);            // [255][20]

  int g2 = blockIdx.x, tid = threadIdx.x;
  int wave = tid >> 5, lane = tid & 31, lr = lane & 15, hi = lane >> 4;

  const unsigned* xg = (const unsigned*)(xbf + (size_t)g2 * 256 * 64);
  for (int i = tid; i < 256 * 64 / 2; i += 256) ((unsigned*)sx)[i] = xg[i];
  __syncthreads();

  // QKV projections: M=o(4 mt), N=n(16 nt), K=c(2 kc)
  for (int proj = 0; proj < 3; ++proj) {
    const unsigned short* Wp = proj == 0 ? Wqp : (proj == 1 ? Wkp : Wvp);
    for (int it = 0; it < 8; ++it) {
      int tt = wave * 8 + it;
      int mt = tt >> 4, nt = tt & 15;
      v8f acc = {};
#pragma unroll
      for (int kc = 0; kc < 2; ++kc) {
        const unsigned short* pa = Wp + (size_t)((mt*2 + kc)*32 + lane) * 16;
        v16u a = ld_frag2(pa, pa + 8);
        const unsigned short* pb = sx + (nt*16 + lr)*64 + kc*32 + (hi ? 16 : 0);
        v16u bb = ld_frag2(pb, pb + 8);
        acc = wmma_bf16(a, bb, acc);
      }
#pragma unroll
      for (int r = 0; r < 8; ++r) {
        int o = mt*16 + r + (hi ? 8 : 0);
        int n = nt*16 + lr;
        float v = acc[r];
        if (proj == 0)      qA[n*64 + o] = f2bf(v * 0.125f);   // /sqrt(64) folded
        else if (proj == 1) kB[n*64 + o] = f2bf(v);
        else                vT[o*256 + n] = f2bf(v);
      }
    }
    __syncthreads();
  }

  for (int tile = 0; tile < 16; ++tile) {
    // logits tile (16 x 256): A = q rows, B = kB. 16 ntiles over 8 waves.
    for (int it = 0; it < 2; ++it) {
      int nt = wave * 2 + it;
      v8f acc = {};
#pragma unroll
      for (int kc = 0; kc < 2; ++kc) {
        const unsigned short* pa = qA + (tile*16 + lr)*64 + kc*32 + (hi ? 16 : 0);
        v16u a = ld_frag2(pa, pa + 8);                     // unpacked A rows
        const unsigned short* pb = kB + (nt*16 + lr)*64 + kc*32 + (hi ? 16 : 0);
        v16u bb = ld_frag2(pb, pb + 8);
        acc = wmma_bf16(a, bb, acc);
      }
#pragma unroll
      for (int r = 0; r < 8; ++r)
        lg[(r + (hi ? 8 : 0))*256 + nt*16 + lr] = acc[r];
    }
    __syncthreads();
    // softmax + top-20 (one thread per row; col 255 is pad -> masked)
    if (tid < 16) {
      int r = tid, ng = tile*16 + r;
      if (ng < 255) {
        float mx = -3.4e38f;
        for (int m = 0; m < 255; ++m) mx = fmaxf(mx, lg[r*256 + m]);
        float tv[20]; short ti[20];
#pragma unroll
        for (int j = 0; j < 20; ++j) { tv[j] = -3.4e38f; ti[j] = 0; }
        float s = 0.f;
        for (int m = 0; m < 255; ++m) {
          float l = lg[r*256 + m];
          float p = __expf(l - mx);
          s += p; lg[r*256 + m] = p;
          if (l > tv[19]) {
            int pos = 19;
            while (pos > 0 && l > tv[pos-1]) { tv[pos] = tv[pos-1]; ti[pos] = ti[pos-1]; --pos; }
            tv[pos] = l; ti[pos] = (short)m;
          }
        }
        float inv = 1.f / s;
        for (int m = 0; m < 255; ++m) pr[r*256 + m] = f2bf(lg[r*256 + m] * inv);
        pr[r*256 + 255] = 0;
#pragma unroll
        for (int j = 0; j < 20; ++j) sidx[ng*20 + j] = ti[j];
      } else {
        for (int m = 0; m < 256; ++m) pr[r*256 + m] = 0;
      }
    }
    __syncthreads();
    // out tile = probs(16x256) @ v(256x64): A = pr rows, B = vT. 4 c-tiles.
    if (wave < 4) {
      int nt = wave;
      v8f acc = {};
#pragma unroll
      for (int kc = 0; kc < 8; ++kc) {
        const unsigned short* pa = pr + lr*256 + kc*32 + (hi ? 16 : 0);
        v16u a = ld_frag2(pa, pa + 8);
        const unsigned short* pb = vT + (nt*16 + lr)*256 + kc*32 + (hi ? 16 : 0);
        v16u bb = ld_frag2(pb, pb + 8);
        acc = wmma_bf16(a, bb, acc);
      }
#pragma unroll
      for (int r = 0; r < 8; ++r) {
        int ng = tile*16 + r + (hi ? 8 : 0);
        if (ng < 255) {
          int cc = nt*16 + lr;
          xt[ng*64 + cc] = f2bf(bf2f(sx[ng*64 + cc]) + acc[r]);
        }
      }
    }
    __syncthreads();
  }

  // edge features: [feat-ctr | ctr] -> y2[g2][n][kk][c128]
  for (int pair = tid; pair < 255 * 20; pair += 256) {
    int n = pair / 20, kk = pair - n * 20;
    int j = sidx[n*20 + kk];
    unsigned short* dst = y2 + (((size_t)g2 * 255 + n) * 20 + kk) * 128;
    for (int c = 0; c < 64; ++c) {
      float fc = bf2f(xt[j*64 + c]);
      float cv = bf2f(xt[n*64 + c]);
      dst[c]      = f2bf(fc - cv);
      dst[64 + c] = f2bf(cv);
    }
  }
}

// ---------------------------------------------------------------------------
// conv2: 1x1 conv as GEMM, M=128 (8 mt), K=128 (4 kc), B read straight from y2.
// ---------------------------------------------------------------------------
__global__ __launch_bounds__(256) void k_conv2(
    const unsigned short* __restrict__ y2, const unsigned short* __restrict__ Wp,
    const float* __restrict__ g2v, const float* __restrict__ b2v,
    float* __restrict__ act2) {
  int tid = threadIdx.x, wave = tid >> 5, lane = tid & 31, lr = lane & 15, hi = lane >> 4;
  int Pb = blockIdx.x * 64;
  int mt = wave;
  float rs = rsqrtf(1.00001f);
  for (int nt = 0; nt < 4; ++nt) {
    int p = Pb + nt*16 + lr;
    v8f acc = {};
#pragma unroll
    for (int kc = 0; kc < 4; ++kc) {
      const unsigned short* pa = Wp + (size_t)((mt*4 + kc)*32 + lane) * 16;
      v16u a = ld_frag2(pa, pa + 8);
      const unsigned short* pb = y2 + (size_t)p * 128 + kc*32 + (hi ? 16 : 0);
      __builtin_prefetch(pb + 128, 0, 1);
      v16u bb = ld_frag2(pb, pb + 8);
      acc = wmma_bf16(a, bb, acc);
    }
#pragma unroll
    for (int r = 0; r < 8; ++r) {
      int o = mt*16 + r + (hi ? 8 : 0);
      float val = acc[r] * (g2v[o] * rs) + b2v[o];
      val = val >= 0.f ? val : 0.2f * val;
      act2[(size_t)o * P2TOT + p] = val;
    }
  }
}

__global__ void k_max2(const float* __restrict__ act2, unsigned short* __restrict__ catb) {
  int gid = blockIdx.x * 256 + threadIdx.x;
  if (gid >= 128 * 64 * 255) return;
  int n = gid % 255; int t = gid / 255;
  int g2 = t & 63, o = t >> 6;
  size_t base = (size_t)o * P2TOT + ((size_t)g2 * 255 + n) * 20;
  float m = -3.4e38f;
#pragma unroll
  for (int j = 0; j < 20; ++j) m = fmaxf(m, act2[base + j]);
  catb[((size_t)g2 * 255 + n) * 192 + 64 + o] = f2bf(m);
}

// ---------------------------------------------------------------------------
// conv5: M=1024 (64 mt), K=192 (6 kc) + bn + lrelu + max over N' via
// order-preserving float->uint keys and atomicMax into emb_u[g2][1024].
// ---------------------------------------------------------------------------
__global__ void k_init_emb(unsigned* __restrict__ embu) {
  int i = blockIdx.x * 256 + threadIdx.x;
  if (i < 65536) embu[i] = 0u;
}

__global__ __launch_bounds__(256) void k_conv5(
    const unsigned short* __restrict__ catb, const unsigned short* __restrict__ Wp,
    const float* __restrict__ g5, const float* __restrict__ b5,
    unsigned* __restrict__ embu) {
  int tid = threadIdx.x, wave = tid >> 5, lane = tid & 31, lr = lane & 15, hi = lane >> 4;
  int Pb = blockIdx.x * 64;
  float rs = rsqrtf(1.00001f);
  for (int it = 0; it < 32; ++it) {
    int idx = wave * 32 + it;
    int mt = idx >> 2, nt = idx & 3;
    int p = Pb + nt*16 + lr;
    v8f acc = {};
#pragma unroll
    for (int kc = 0; kc < 6; ++kc) {
      const unsigned short* pa = Wp + (size_t)((mt*6 + kc)*32 + lane) * 16;
      v16u a = ld_frag2(pa, pa + 8);
      const unsigned short* pb = catb + (size_t)p * 192 + kc*32 + (hi ? 16 : 0);
      v16u bb = ld_frag2(pb, pb + 8);
      acc = wmma_bf16(a, bb, acc);
    }
    int g2 = p / 255;
#pragma unroll
    for (int r = 0; r < 8; ++r) {
      int e = mt*16 + r + (hi ? 8 : 0);
      float val = acc[r] * (g5[e] * rs) + b5[e];
      val = val >= 0.f ? val : 0.2f * val;
      unsigned u = __builtin_bit_cast(unsigned, val);
      unsigned key = (u & 0x80000000u) ? ~u : (u | 0x80000000u);
      atomicMax(&embu[(size_t)g2 * 1024 + e], key);
    }
  }
}

__global__ void k_final(const unsigned* __restrict__ embu, float* __restrict__ out) {
  int i = blockIdx.x * 256 + threadIdx.x;
  if (i >= 65536) return;
  unsigned key = embu[i];
  unsigned u = (key & 0x80000000u) ? (key ^ 0x80000000u) : ~key;
  out[i] = __builtin_bit_cast(float, u);   // out layout (B,F',EMB) == [g2][e]
}

// ---------------------------------------------------------------------------
// Workspace layout (all 256B aligned)
// ---------------------------------------------------------------------------
static constexpr size_t OFF_Y1   = 0;                      // bf16 8*6*32*512*20
static constexpr size_t OFF_WP1  = OFF_Y1   + 31457280;    // 4*7*32*16 bf16
static constexpr size_t OFF_WP2  = OFF_WP1  + 28672;       // 8*4*32*16
static constexpr size_t OFF_WP5  = OFF_WP2  + 32768;       // 64*6*32*16
static constexpr size_t OFF_WQP  = OFF_WP5  + 393216;      // 4*2*32*16
static constexpr size_t OFF_WKP  = OFF_WQP  + 8192;
static constexpr size_t OFF_WVP  = OFF_WKP  + 8192;
static constexpr size_t OFF_ACT1 = OFF_WVP  + 8192;        // f32 64*293760
static constexpr size_t OFF_XBF  = OFF_ACT1 + 75202560;    // bf16 64*256*64
static constexpr size_t OFF_CAT  = OFF_XBF  + 2097152;     // bf16 16320*192
static constexpr size_t OFF_Y2   = OFF_CAT  + 6266880;     // bf16 64*255*20*128
static constexpr size_t OFF_ACT2 = OFF_Y2   + 83558400;    // f32 128*326400
static constexpr size_t OFF_EMBU = OFF_ACT2 + 167116800;   // u32 65536

extern "C" void kernel_launch(void* const* d_in, const int* in_sizes, int n_in,
                              void* d_out, int out_size, void* d_ws, size_t ws_size,
                              hipStream_t stream) {
  (void)in_sizes; (void)n_in; (void)out_size; (void)ws_size;
  const float* x      = (const float*)d_in[0];
  const float* Wq1    = (const float*)d_in[1];
  const float* Wk1    = (const float*)d_in[2];
  const float* Wv1    = (const float*)d_in[3];
  const float* Wq2    = (const float*)d_in[4];
  const float* Wk2    = (const float*)d_in[5];
  const float* Wv2    = (const float*)d_in[6];
  const float* conv1w = (const float*)d_in[7];
  const float* conv2w = (const float*)d_in[8];
  const float* conv5w = (const float*)d_in[9];
  const float* bn1g = (const float*)d_in[10];
  const float* bn1b = (const float*)d_in[11];
  const float* bn2g = (const float*)d_in[12];
  const float* bn2b = (const float*)d_in[13];
  const float* bn5g = (const float*)d_in[14];
  const float* bn5b = (const float*)d_in[15];

  char* ws = (char*)d_ws;
  unsigned short* y1  = (unsigned short*)(ws + OFF_Y1);
  unsigned short* wp1 = (unsigned short*)(ws + OFF_WP1);
  unsigned short* wp2 = (unsigned short*)(ws + OFF_WP2);
  unsigned short* wp5 = (unsigned short*)(ws + OFF_WP5);
  unsigned short* wqp = (unsigned short*)(ws + OFF_WQP);
  unsigned short* wkp = (unsigned short*)(ws + OFF_WKP);
  unsigned short* wvp = (unsigned short*)(ws + OFF_WVP);
  float*          a1  = (float*)(ws + OFF_ACT1);
  unsigned short* xbf = (unsigned short*)(ws + OFF_XBF);
  unsigned short* cb  = (unsigned short*)(ws + OFF_CAT);
  unsigned short* y2  = (unsigned short*)(ws + OFF_Y2);
  float*          a2  = (float*)(ws + OFF_ACT2);
  unsigned*       eu  = (unsigned*)(ws + OFF_EMBU);

  // weight repack to bf16 A-fragment order
  k_pack<<<4,  256, 0, stream>>>(conv1w, wp1, 64,   216, 7);
  k_pack<<<4,  256, 0, stream>>>(conv2w, wp2, 128,  128, 4);
  k_pack<<<48, 256, 0, stream>>>(conv5w, wp5, 1024, 192, 6);
  k_pack<<<1,  256, 0, stream>>>(Wq2,    wqp, 64,   64,  2);
  k_pack<<<1,  256, 0, stream>>>(Wk2,    wkp, 64,   64,  2);
  k_pack<<<1,  256, 0, stream>>>(Wv2,    wvp, 64,   64,  2);

  k_stage1<<<256, 256, 0, stream>>>(x, Wq1, Wk1, Wv1, y1);
  k_conv1<<<4590, 128, 0, stream>>>(y1, wp1, bn1g, bn1b, a1);
  k_max1<<<4080, 256, 0, stream>>>(a1, xbf, cb);
  k_stage2<<<64, 256, 198656, stream>>>(xbf, wqp, wkp, wvp, y2);
  k_conv2<<<5100, 256, 0, stream>>>(y2, wp2, bn2g, bn2b, a2);
  k_max2<<<8160, 256, 0, stream>>>(a2, cb);
  k_init_emb<<<256, 256, 0, stream>>>(eu);
  k_conv5<<<255, 256, 0, stream>>>(cb, wp5, bn5g, bn5b, eu);
  k_final<<<256, 256, 0, stream>>>(eu, (float*)d_out);
}
